// NetGIN_70677981823696
// MI455X (gfx1250) — matrix operand order repaired and checked by reading, not compiled, not run on hardware
//
#include <hip/hip_runtime.h>
#include <math.h>

#define DH 128          // hidden dim
#define BM 128          // GEMM row tile
#define BK 32           // GEMM k tile
#define BNC 128         // GEMM col tile (always 128)
#define SAPAD 40        // sA row stride (elements): 80B, 16B-aligned, bank-spread
#define SBPAD 40        // sB col stride (elements)

typedef __bf16 bf16t;
typedef __bf16 v16bf __attribute__((ext_vector_type(16)));
typedef __bf16 v8bf  __attribute__((ext_vector_type(8)));
typedef float  v8f   __attribute__((ext_vector_type(8)));

union V16U { v16bf v; v8bf h[2]; };

__device__ __forceinline__ unsigned bfbits(float f) {
    unsigned u = __float_as_uint(f);
    return (u + 0x7FFFu + ((u >> 16) & 1u)) >> 16;   // RNE truncate to bf16
}
__device__ __forceinline__ unsigned pack_bf2(float lo, float hi) {
    return (bfbits(hi) << 16) | bfbits(lo);
}
__device__ __forceinline__ float sigmoidf_(float x) { return 1.f / (1.f + expf(-x)); }

// monotone float->uint map for atomicMax-based segment max
__device__ __forceinline__ unsigned fmap(float f) {
    unsigned u = __float_as_uint(f);
    return (u & 0x80000000u) ? ~u : (u | 0x80000000u);
}
__device__ __forceinline__ float funmap(unsigned u) {
    unsigned v = (u & 0x80000000u) ? (u & 0x7FFFFFFFu) : ~u;
    return __uint_as_float(v);
}

// ---------------------------------------------------------------------------
// WMMA bf16 GEMM:  C[M x 128] = A[M x K] (lda) @ W[K x 128] + bias, store ldc
// block: 256 threads = 8 waves; wave w owns rows w*16..w*16+15, all 128 cols
// 8 accumulators (16x16) per wave; fragments read as contiguous b128 LDS loads
// ---------------------------------------------------------------------------
__global__ __launch_bounds__(256)
void gemm_wmma(const float* __restrict__ A, int lda,
               const float* __restrict__ Wm,      // K x 128 row-major
               const float* __restrict__ bias,    // 128
               float* __restrict__ C, int ldc,
               int M, int K)
{
    __shared__ bf16t sA[BM * SAPAD];    // row-major, padded rows   (10 KB)
    __shared__ bf16t sB[BNC * SBPAD];   // col-major, padded cols   (10 KB)

    const int tid  = threadIdx.x;
    const int wave = tid >> 5;     // 0..7 -> row strip
    const int lane = tid & 31;
    const int hf   = lane >> 4;
    const int ml   = lane & 15;
    const long row0 = (long)blockIdx.x * BM;

    v8f acc[8];
#pragma unroll
    for (int t = 0; t < 8; ++t)
#pragma unroll
        for (int e = 0; e < 8; ++e) acc[t][e] = 0.0f;

    for (int k0 = 0; k0 < K; k0 += BK) {
        if (k0 + BK < K) __builtin_prefetch(&Wm[(long)(k0 + BK) * BNC], 0, 0);

        // stage A tile 128x32 as bf16 pairs (natural row-major, padded stride)
        for (int p = tid; p < BM * BK / 2; p += 256) {
            int r = p >> 4, c = (p & 15) * 2;
            long gr = row0 + r; int gc = k0 + c;
            float v0 = (gr < M && gc < K)     ? A[gr * (long)lda + gc]     : 0.0f;
            float v1 = (gr < M && gc + 1 < K) ? A[gr * (long)lda + gc + 1] : 0.0f;
            *(unsigned*)&sA[r * SAPAD + c] = pack_bf2(v0, v1);
        }
        // stage W tile 32x128 transposed into col-major (col stride SBPAD)
        for (int p = tid; p < BK * BNC / 2; p += 256) {
            int kk = (p >> 7) * 2, c = p & 127;
            int g0 = k0 + kk;
            float v0 = (g0 < K)     ? Wm[(long)g0 * BNC + c]       : 0.0f;
            float v1 = (g0 + 1 < K) ? Wm[(long)(g0 + 1) * BNC + c] : 0.0f;
            *(unsigned*)&sB[c * SBPAD + kk] = pack_bf2(v0, v1);
        }
        __syncthreads();

        // A fragment: lane (hf,ml) of row (wave*16+ml): two contiguous 16B runs
        // hf=0 -> kk {0..15} ; hf=1 -> kk {16..31}
        V16U af;
        const bf16t* ap = &sA[(wave * 16 + ml) * SAPAD];
        af.h[0] = *(const v8bf*)(ap + hf * 16);
        af.h[1] = *(const v8bf*)(ap + hf * 16 + 8);

#pragma unroll
        for (int t = 0; t < 8; ++t) {
            // B fragment: col = t*16+ml, kk = hf*16 + e : one 32B contiguous run
            V16U bf_;
            const bf16t* bp = &sB[(t * 16 + ml) * SBPAD + hf * 16];
            bf_.h[0] = *(const v8bf*)(bp);
            bf_.h[1] = *(const v8bf*)(bp + 8);
            acc[t] = __builtin_amdgcn_wmma_f32_16x16x32_bf16(
                false, af.v, false, bf_.v, (short)0, acc[t], false, false);
        }
        __syncthreads();
    }

    // C/D layout: vgpr r -> lanes 0-15: M=r, lanes 16-31: M=r+8 ; N = lane&15
#pragma unroll
    for (int t = 0; t < 8; ++t) {
        int ncol = t * 16 + ml;
        float b = bias[ncol];
#pragma unroll
        for (int r = 0; r < 8; ++r) {
            long mrow = row0 + wave * 16 + r + hf * 8;
            if (mrow < M) C[mrow * (long)ldc + ncol] = acc[t][r] + b;
        }
    }
}

// ---------------------------------------------------------------------------
// BatchNorm helpers
// ---------------------------------------------------------------------------
__global__ void zero_f(float* __restrict__ p, long n) {
    long i = (long)blockIdx.x * blockDim.x + threadIdx.x;
    for (; i < n; i += (long)gridDim.x * blockDim.x) p[i] = 0.f;
}

// 128 threads, each owns one column; block covers a 256-row chunk (coalesced)
__global__ void colstats(const float* __restrict__ H, int M, float* __restrict__ stats) {
    int col = threadIdx.x;
    long r0 = (long)blockIdx.x * 256;
    long r1 = r0 + 256; if (r1 > M) r1 = M;
    float s = 0.f, q = 0.f;
    for (long r = r0; r < r1; ++r) {
        float v = H[r * DH + col];
        s += v; q += v * v;
    }
    atomicAdd(&stats[col], s);
    atomicAdd(&stats[DH + col], q);
}

__global__ void bnrelu(const float* __restrict__ H, const float* __restrict__ stats,
                       const float* __restrict__ g, const float* __restrict__ be,
                       float invM, float* __restrict__ Dst, int dstStride) {
    int col = threadIdx.x;
    long row = blockIdx.x;
    float m   = stats[col] * invM;
    float var = stats[DH + col] * invM - m * m;
    float v = H[row * DH + col];
    float o = g[col] * (v - m) * rsqrtf(var + 1e-5f) + be[col];
    Dst[row * (long)dstStride + col] = fmaxf(o, 0.f);
}

// ---------------------------------------------------------------------------
// input concat builders
// ---------------------------------------------------------------------------
__global__ void build_na(const float* __restrict__ a, const float* __restrict__ b,
                         float* __restrict__ out, int n) {
    long i = (long)blockIdx.x * blockDim.x + threadIdx.x;
    long tot = (long)n * 26;
    if (i >= tot) return;
    long r = i / 26; int c = (int)(i % 26);
    out[i] = (c < 13) ? a[r * 13 + c] : b[r * 13 + (c - 13)];
}
__global__ void build_ee(const float* __restrict__ ea, const float* __restrict__ dist,
                         float* __restrict__ out, int n) {
    long i = (long)blockIdx.x * blockDim.x + threadIdx.x;
    long tot = (long)n * 5;
    if (i >= tot) return;
    long r = i / 5; int c = (int)(i % 5);
    out[i] = (c < 4) ? ea[r * 4 + c] : dist[r];
}

// ---------------------------------------------------------------------------
// GIN aggregation
// ---------------------------------------------------------------------------
__global__ void scale_copy(float* __restrict__ z, const float* __restrict__ x,
                           const float* __restrict__ geps, int idx, long n) {
    long i = (long)blockIdx.x * blockDim.x + threadIdx.x;
    if (i >= n) return;
    float s = 1.0f + geps[idx];
    z[i] = s * x[i];
}
__global__ void scatter_edges(float* __restrict__ z, const float* __restrict__ x,
                              const int* __restrict__ ei, int ne) {
    const int* src = ei;
    const int* dst = ei + ne;
    int col = threadIdx.x; // 128
    for (long e = blockIdx.x; e < ne; e += gridDim.x) {
        long s = src[e], d = dst[e];
        atomicAdd(&z[d * DH + col], x[s * DH + col]);
    }
}

// ---------------------------------------------------------------------------
// Set2Set
// ---------------------------------------------------------------------------
__global__ void lstm_step(const float* __restrict__ qs, float* __restrict__ h,
                          float* __restrict__ c,
                          const float* __restrict__ Wih,  // 512 x 256
                          const float* __restrict__ Whh,  // 512 x 128
                          const float* __restrict__ bih, const float* __restrict__ bhh) {
    int g = blockIdx.x, d = threadIdx.x;
    __shared__ float q[256];
    __shared__ float hh[DH];
    q[d] = qs[(long)g * 256 + d];
    q[DH + d] = qs[(long)g * 256 + DH + d];
    hh[d] = h[(long)g * DH + d];
    __syncthreads();
    float z[4];
#pragma unroll
    for (int gate = 0; gate < 4; ++gate) {
        int j = gate * DH + d;
        float s = bih[j] + bhh[j];
        const float* wr = &Wih[(long)j * 256];
        for (int k = 0; k < 256; ++k) s += q[k] * wr[k];
        const float* wh = &Whh[(long)j * DH];
        for (int k = 0; k < DH; ++k) s += hh[k] * wh[k];
        z[gate] = s;
    }
    float cn = sigmoidf_(z[1]) * c[(long)g * DH + d] + sigmoidf_(z[0]) * tanhf(z[2]);
    float hn = sigmoidf_(z[3]) * tanhf(cn);
    c[(long)g * DH + d] = cn;
    h[(long)g * DH + d] = hn;
}

// one wave per row: e[row] = dot(x[row], h[batch[row]])
__global__ void attn_scores(const float* __restrict__ x, const float* __restrict__ h,
                            const int* __restrict__ batch, float* __restrict__ e, int n) {
    int wave = threadIdx.x >> 5, lane = threadIdx.x & 31;
    long row = (long)blockIdx.x * 8 + wave;
    if (row >= n) return;
    int g = batch[row];
    float s = 0.f;
    for (int k = lane; k < DH; k += 32) s += x[row * DH + k] * h[(long)g * DH + k];
#pragma unroll
    for (int o = 16; o > 0; o >>= 1) s += __shfl_xor(s, o);
    if (lane == 0) e[row] = s;
}

__global__ void seg_init(unsigned* __restrict__ emaxu, float* __restrict__ den,
                         float* __restrict__ rb) {
    int g = blockIdx.x, d = threadIdx.x;
    rb[(long)g * DH + d] = 0.f;
    if (d == 0) { emaxu[g] = fmap(-3.402823466e38f); den[g] = 0.f; }
}
__global__ void segmax_k(const float* __restrict__ e, const int* __restrict__ batch,
                         unsigned* __restrict__ emaxu, int n) {
    long i = (long)blockIdx.x * blockDim.x + threadIdx.x;
    if (i < n) atomicMax(&emaxu[batch[i]], fmap(e[i]));
}
__global__ void expw_k(const float* __restrict__ e, const int* __restrict__ batch,
                       const unsigned* __restrict__ emaxu, float* __restrict__ w,
                       float* __restrict__ den, int n) {
    long i = (long)blockIdx.x * blockDim.x + threadIdx.x;
    if (i >= n) return;
    int g = batch[i];
    float ww = expf(e[i] - funmap(emaxu[g]));
    w[i] = ww;
    atomicAdd(&den[g], ww);
}
__global__ void raccum(const float* __restrict__ x, const float* __restrict__ w,
                       const float* __restrict__ den, const int* __restrict__ batch,
                       float* __restrict__ rb) {
    long row = blockIdx.x; int col = threadIdx.x;
    int g = batch[row];
    float a = w[row] / den[g];
    atomicAdd(&rb[(long)g * DH + col], a * x[row * DH + col]);
}
__global__ void pack_q(float* __restrict__ qs, const float* __restrict__ h,
                       const float* __restrict__ rb) {
    int g = blockIdx.x, d = threadIdx.x;
    qs[(long)g * 256 + d] = h[(long)g * DH + d];
    qs[(long)g * 256 + DH + d] = rb[(long)g * DH + d];
}

// out = relu(qs @ fc1W + fc1b) @ fc4W + fc4b
__global__ void head_k(const float* __restrict__ qs, const float* __restrict__ fc1W,
                       const float* __restrict__ fc1b, const float* __restrict__ fc4W,
                       const float* __restrict__ fc4b, float* __restrict__ out) {
    int g = blockIdx.x, d = threadIdx.x;
    __shared__ float q[256];
    __shared__ float t[DH];
    q[d] = qs[(long)g * 256 + d];
    q[DH + d] = qs[(long)g * 256 + DH + d];
    __syncthreads();
    float s = fc1b[d];
    for (int k = 0; k < 256; ++k) s += q[k] * fc1W[(long)k * DH + d];
    t[d] = fmaxf(s, 0.f);
    __syncthreads();
    if (d < 12) {
        float s2 = fc4b[d];
        for (int k = 0; k < DH; ++k) s2 += t[k] * fc4W[k * 12 + d];
        out[(long)g * 12 + d] = s2;
    }
}

// ---------------------------------------------------------------------------
extern "C" void kernel_launch(void* const* d_in, const int* in_sizes, int n_in,
                              void* d_out, int out_size, void* d_ws, size_t ws_size,
                              hipStream_t stream) {
    const int N = in_sizes[0] / 3;
    const int E = in_sizes[62] / 2;
    const int G = out_size / 12;

    const float* x_type    = (const float*)d_in[0];
    const float* first     = (const float*)d_in[1];
    const float* second    = (const float*)d_in[2];
    const float* edge_attr = (const float*)d_in[3];
    const float* dist      = (const float*)d_in[4];
    // encoder blocks: base index + {W1,b1,g1,be1,W2,b2,g2,be2}
    const float* enc[4][8];
    for (int b = 0; b < 4; ++b)
        for (int j = 0; j < 8; ++j) enc[b][j] = (const float*)d_in[5 + b * 8 + j];
    const float* gW1  = (const float*)d_in[37];
    const float* gb1  = (const float*)d_in[38];
    const float* gg1  = (const float*)d_in[39];
    const float* gbe1 = (const float*)d_in[40];
    const float* gW2  = (const float*)d_in[41];
    const float* gb2  = (const float*)d_in[42];
    const float* gg2  = (const float*)d_in[43];
    const float* gbe2 = (const float*)d_in[44];
    const float* geps = (const float*)d_in[45];
    const float* mW1  = (const float*)d_in[46];
    const float* mb1  = (const float*)d_in[47];
    const float* mg1  = (const float*)d_in[48];
    const float* mbe1 = (const float*)d_in[49];
    const float* mW2  = (const float*)d_in[50];
    const float* mb2  = (const float*)d_in[51];
    const float* mg2  = (const float*)d_in[52];
    const float* mbe2 = (const float*)d_in[53];
    const float* Wih  = (const float*)d_in[54];
    const float* Whh  = (const float*)d_in[55];
    const float* bih  = (const float*)d_in[56];
    const float* bhh  = (const float*)d_in[57];
    const float* fc1W = (const float*)d_in[58];
    const float* fc1b = (const float*)d_in[59];
    const float* fc4W = (const float*)d_in[60];
    const float* fc4b = (const float*)d_in[61];
    const int* eis[4] = {(const int*)d_in[62], (const int*)d_in[63],
                         (const int*)d_in[64], (const int*)d_in[65]};
    const int* batch = (const int*)d_in[66];

    // workspace carve-up
    float* ws = (float*)d_ws;
    size_t off = 0;
    float* cat = ws + off; off += (size_t)N * 512;  // concat buf (384 stride in encoder phase)
    float* h1  = ws + off; off += (size_t)N * DH;
    float* h2  = ws + off; off += (size_t)N * DH;
    float* zb  = ws + off; off += (size_t)N * DH;
    float* xb  = ws + off; off += (size_t)N * DH;
    float* ev  = ws + off; off += (size_t)N;
    float* wv  = ws + off; off += (size_t)N;
    float* stats = ws + off; off += 256;
    float* qs  = ws + off; off += (size_t)G * 256;
    float* hb  = ws + off; off += (size_t)G * DH;
    float* cb  = ws + off; off += (size_t)G * DH;
    float* rb  = ws + off; off += (size_t)G * DH;
    unsigned* emaxu = (unsigned*)(ws + off); off += G;
    float* den = ws + off; off += G;
    float* in_na = zb;                     // aliases zb (free during encoder phase)
    float* in_ee = zb + (size_t)N * 26;

    auto gemm = [&](const float* A, int lda, const float* Wm, const float* bias,
                    float* C, int ldc, int M, int K) {
        gemm_wmma<<<dim3((M + BM - 1) / BM), 256, 0, stream>>>(A, lda, Wm, bias, C, ldc, M, K);
    };
    auto bn = [&](const float* H, const float* g, const float* be, float* Dst, int ldc, int M) {
        zero_f<<<1, 256, 0, stream>>>(stats, 256);
        colstats<<<(M + 255) / 256, DH, 0, stream>>>(H, M, stats);
        bnrelu<<<M, DH, 0, stream>>>(H, stats, g, be, 1.0f / (float)M, Dst, ldc);
    };

    // ---- encoders ----
    {
        long t26 = (long)N * 26, t5 = (long)N * 5;
        build_na<<<(t26 + 255) / 256, 256, 0, stream>>>(first, second, in_na, N);
        build_ee<<<(t5 + 255) / 256, 256, 0, stream>>>(edge_attr, dist, in_ee, N);
    }
    const float* encA[3] = {x_type, in_na, in_ee};
    const int    encK[3] = {3, 26, 5};
    for (int b = 0; b < 3; ++b) {
        gemm(encA[b], encK[b], enc[b][0], enc[b][1], h1, DH, N, encK[b]);
        bn(h1, enc[b][2], enc[b][3], h2, DH, N);
        gemm(h2, DH, enc[b][4], enc[b][5], h1, DH, N, DH);
        bn(h1, enc[b][6], enc[b][7], cat + b * DH, 384, N);
    }
    // m0 fusion: cat (N x 384) -> x
    gemm(cat, 384, enc[3][0], enc[3][1], h1, DH, N, 384);
    bn(h1, enc[3][2], enc[3][3], h2, DH, N);
    gemm(h2, DH, enc[3][4], enc[3][5], h1, DH, N, DH);
    bn(h1, enc[3][6], enc[3][7], xb, DH, N);

    // ---- 6 GIN layers ----
    const int catoff[4] = {0, 256, 128, 384};  // cat order [x1, x3, x2, x4]
    const long n128 = (long)N * DH;
    for (int L = 0; L < 6; ++L) {
        for (int k = 0; k < 4; ++k) {
            int i = 4 * L + k;
            scale_copy<<<(n128 + 255) / 256, 256, 0, stream>>>(zb, xb, geps, i, n128);
            scatter_edges<<<16384, DH, 0, stream>>>(zb, xb, eis[k], E);
            gemm(zb, DH, gW1 + (size_t)i * DH * DH, gb1 + (size_t)i * DH, h1, DH, N, DH);
            bn(h1, gg1 + (size_t)i * DH, gbe1 + (size_t)i * DH, h2, DH, N);
            gemm(h2, DH, gW2 + (size_t)i * DH * DH, gb2 + (size_t)i * DH, h1, DH, N, DH);
            bn(h1, gg2 + (size_t)i * DH, gbe2 + (size_t)i * DH, cat + catoff[k], 512, N);
        }
        gemm(cat, 512, mW1 + (size_t)L * 512 * DH, mb1 + (size_t)L * DH, h1, DH, N, 512);
        bn(h1, mg1 + (size_t)L * DH, mbe1 + (size_t)L * DH, h2, DH, N);
        gemm(h2, DH, mW2 + (size_t)L * DH * DH, mb2 + (size_t)L * DH, h1, DH, N, DH);
        bn(h1, mg2 + (size_t)L * DH, mbe2 + (size_t)L * DH, xb, DH, N);
    }

    // ---- Set2Set (6 steps) ----
    zero_f<<<64, 256, 0, stream>>>(qs, (long)G * 256);
    zero_f<<<64, 256, 0, stream>>>(hb, (long)G * DH);
    zero_f<<<64, 256, 0, stream>>>(cb, (long)G * DH);
    for (int s = 0; s < 6; ++s) {
        lstm_step<<<G, DH, 0, stream>>>(qs, hb, cb, Wih, Whh, bih, bhh);
        attn_scores<<<(N + 7) / 8, 256, 0, stream>>>(xb, hb, batch, ev, N);
        seg_init<<<G, DH, 0, stream>>>(emaxu, den, rb);
        segmax_k<<<(N + 255) / 256, 256, 0, stream>>>(ev, batch, emaxu, N);
        expw_k<<<(N + 255) / 256, 256, 0, stream>>>(ev, batch, emaxu, wv, den, N);
        raccum<<<N, DH, 0, stream>>>(xb, wv, den, batch, rb);
        pack_q<<<G, DH, 0, stream>>>(qs, hb, rb);
    }

    head_k<<<G, DH, 0, stream>>>(qs, fc1W, fc1b, fc4W, fc4b, (float*)d_out);
}
